// ONINorm_colum_44676249813435
// MI455X (gfx1250) — compile-verified, matrix operand order checked by reading, not run on hardware
//
#include <hip/hip_runtime.h>
#include <hip/hip_bf16.h>

typedef __attribute__((ext_vector_type(16))) _Float16 v16h;
typedef __attribute__((ext_vector_type(8)))  float    v8f;

#define LDS_STRIDE 40   // halves per LDS tile row (32 used + 8 pad, 16B-aligned rows)

// Fragment (lane,e) -> K mapping per CDNA5 ISA 7.12.2 (16-bit A-matrix 16x32):
// lane L: row = base + (L&15), h = L>>4; e<8 -> k=8h+e ; e>=8 -> k=16+8h+(e-8)

// Gather a 16x32 f16 fragment from an LDS tile stored [row][LDS_STRIDE]
__device__ __forceinline__ v16h load_frag(const _Float16* tile, int rowBase) {
  const int lane = threadIdx.x & 31;
  const int r = lane & 15, h = lane >> 4;
  const int* ip = (const int*)(tile + (size_t)(rowBase + r) * LDS_STRIDE);
  union { v16h v; int i[8]; } f;
  const int d0 = 4 * h;
#pragma unroll
  for (int j = 0; j < 4; ++j) {
    f.i[j]     = ip[d0 + j];
    f.i[4 + j] = ip[d0 + 8 + j];
  }
  return f.v;
}

// Load a pre-packed fragment: 16 contiguous halves per lane at block + lane*16
__device__ __forceinline__ v16h load_packed(const _Float16* blk) {
  const int lane = threadIdx.x & 31;
  union { v16h v; int4 q[2]; } f;
  const int4* p = (const int4*)(blk + lane * 16);
  f.q[0] = p[0];
  f.q[1] = p[1];
  return f.v;
}

// ---------------------------------------------------------------------------
// Pack Zc (natural [c][f], ld=512, 4 groups of 16384 rows) into WMMA-fragment
// "B-form": per group, block (fT, cT) [fT<32, cT<512] holds 512 halves:
// lane L stores halves e=0..15 with f = fT*16+(L&15), c = cT*32 + k(L,e).
// This same buffer serves as the A-operand of Zc^T (packed-A(Zc^T)==packed-B(Zc)).
// LDS-tiled 64c x 64f transpose -> one output block per wave.
// ---------------------------------------------------------------------------
__global__ __launch_bounds__(256) void pack_zc_bform(
    const _Float16* __restrict__ Zc, _Float16* __restrict__ ZcB) {
  __shared__ _Float16 Ln[64 * 72];  // [c][f], 144B row stride (16B multiple)
  const int bid = blockIdx.x;       // 8192 = 4 groups * 256 cTiles64 * 8 fTiles64
  const int g = bid >> 11;
  const int rem = bid & 2047;
  const int cb = rem >> 3;          // c / 64
  const int fb = rem & 7;           // f / 64
  const _Float16* src = Zc + (size_t)g * (16384 * 512);
  const int t = threadIdx.x;
#pragma unroll
  for (int j = 0; j < 2; ++j) {
    const int ch = t + 256 * j;     // 0..511
    const int c = ch >> 3, f8 = (ch & 7) * 8;
    const int4 v = *(const int4*)(src + (size_t)(cb * 64 + c) * 512 + fb * 64 + f8);
    *(int4*)(Ln + c * 72 + f8) = v;
  }
  __syncthreads();
  const int wave = t >> 5, lane = t & 31;
  const int fTl = wave >> 1, cTl = wave & 1;   // f-tile16 0..3, c-tile32 0..1
  const int fT = fb * 4 + fTl;                 // global f/16 tile  (0..31)
  const int cT = cb * 2 + cTl;                 // global c/32 tile  (0..511)
  const int fl = fTl * 16 + (lane & 15);
  const int h = lane >> 4;
  union { int4 q[2]; _Float16 hv[16]; } o;
#pragma unroll
  for (int e = 0; e < 16; ++e) {
    const int cl = (e < 8) ? (8 * h + e) : (16 + 8 * h + (e - 8));
    o.hv[e] = Ln[(cTl * 32 + cl) * 72 + fl];
  }
  _Float16* dst = ZcB + (size_t)g * (16384 * 512) +
                  ((size_t)fT * 512 + cT) * 512 + lane * 16;
  ((int4*)dst)[0] = o.q[0];
  ((int4*)dst)[1] = o.q[1];
}

// ---------------------------------------------------------------------------
// S = Zc^T Zc from packed fragments: both operands stream from ZcB.
// Block = 256 threads (8 waves), 128x128 output tile, K-slice = 32.
// Wave w: 64x32 macro-tile: row-tiles rt..rt+3 (rt=(w>>2)*4), col-tiles
// ct..ct+1 (ct=(w&3)*2) -> 8 WMMAs / slice. All staging & fragment loads are
// vector b128 ops (no scalar LDS traffic). Global prefetch double-buffers.
// ---------------------------------------------------------------------------
__global__ __launch_bounds__(256) void syrk_packed(
    const _Float16* __restrict__ P, float* __restrict__ S) {
  __shared__ alignas(16) _Float16 Asp[8 * 512];
  __shared__ alignas(16) _Float16 Bsp[8 * 512];
  const int g = blockIdx.z;
  const _Float16* Pg = P + (size_t)g * (16384 * 512);
  float* Sg = S + (size_t)g * (512 * 512);
  const int t = threadIdx.x, lane = t & 31, wave = t >> 5;
  const int tm0 = blockIdx.y * 8, tn0 = blockIdx.x * 8;   // in 16-tiles
  const int rt = (wave >> 2) * 4, ct = (wave & 3) * 2;
  const int blkL = t >> 5, inner = t & 31;                // staging role

  v8f acc[4][2];
#pragma unroll
  for (int i = 0; i < 4; ++i)
#pragma unroll
    for (int j = 0; j < 2; ++j)
#pragma unroll
      for (int e = 0; e < 8; ++e) acc[i][j][e] = 0.0f;

  int4 ra[2], rb[2];
  auto load_global = [&](int tk) {
    const int4* pa = (const int4*)(Pg + ((size_t)(tm0 + blkL) * 512 + tk) * 512 + inner * 16);
    const int4* pb = (const int4*)(Pg + ((size_t)(tn0 + blkL) * 512 + tk) * 512 + inner * 16);
    ra[0] = pa[0]; ra[1] = pa[1];
    rb[0] = pb[0]; rb[1] = pb[1];
  };
  auto store_lds = [&]() {
    int4* da = (int4*)(Asp + blkL * 512 + inner * 16);
    int4* db = (int4*)(Bsp + blkL * 512 + inner * 16);
    da[0] = ra[0]; da[1] = ra[1];
    db[0] = rb[0]; db[1] = rb[1];
  };

  load_global(0);
  for (int tk = 0; tk < 512; ++tk) {
    store_lds();
    __syncthreads();
    if (tk + 1 < 512) load_global(tk + 1);

    v16h a[4], b[2];
#pragma unroll
    for (int i = 0; i < 4; ++i) a[i] = load_packed(Asp + (rt + i) * 512);
#pragma unroll
    for (int j = 0; j < 2; ++j) b[j] = load_packed(Bsp + (ct + j) * 512);
#pragma unroll
    for (int i = 0; i < 4; ++i)
#pragma unroll
      for (int j = 0; j < 2; ++j)
        acc[i][j] = __builtin_amdgcn_wmma_f32_16x16x32_f16(
            false, a[i], false, b[j], (short)0, acc[i][j], false, false);
    __syncthreads();
  }

  const int hh = (lane >> 4) * 8;
#pragma unroll
  for (int i = 0; i < 4; ++i)
#pragma unroll
    for (int j = 0; j < 2; ++j) {
      const int mbase = (tm0 + rt + i) * 16 + hh;
      const int nn = (tn0 + ct + j) * 16 + (lane & 15);
#pragma unroll
      for (int r = 0; r < 8; ++r)
        Sg[(size_t)(mbase + r) * 512 + nn] = acc[i][j][r];
    }
}

// ---------------------------------------------------------------------------
// Generic batched 64x64-tile GEMM (natural layouts), K-slice 32, prefetch.
// Wave w: tiles (tmA, tn) and (tmA+2, tn), tn=w&3, tmA=w>>2. Used for the
// small Newton-Schulz chain and the final Zc@B (f32 out, per-group scale).
// ---------------------------------------------------------------------------
template <bool OUT_F32>
__global__ __launch_bounds__(256) void gemm_wmma(
    const _Float16* __restrict__ A, const _Float16* __restrict__ B,
    const _Float16* __restrict__ Cadd, void* __restrict__ Dv,
    int K, long long sA, long long sB, long long sC, long long sD,
    int lda, int ldb, int ldd, float alpha, float beta,
    const float* __restrict__ scale) {
  __shared__ alignas(16) _Float16 As[64 * LDS_STRIDE];
  __shared__ alignas(16) _Float16 Bs[64 * LDS_STRIDE];
  const int g = blockIdx.z;
  A += (size_t)g * sA;
  B += (size_t)g * sB;
  if (Cadd) Cadd += (size_t)g * sC;

  const int t = threadIdx.x;
  const int lane = t & 31, wave = t >> 5;
  const int m0 = blockIdx.y * 64, n0 = blockIdx.x * 64;
  const int tn = wave & 3, tmA = wave >> 2;

  v8f acc0, acc1;
#pragma unroll
  for (int i = 0; i < 8; ++i) { acc0[i] = 0.0f; acc1[i] = 0.0f; }

  int4 ra, rb;
  auto load_global = [&](int kk) {
    const int m = t >> 2, k8 = (t & 3) * 8;
    ra = *(const int4*)(A + (size_t)(m0 + m) * lda + kk + k8);
    const int k = t >> 3, n8 = (t & 7) * 8;
    rb = *(const int4*)(B + (size_t)(kk + k) * ldb + n0 + n8);
  };
  auto store_lds = [&]() {
    const int m = t >> 2, k8 = (t & 3) * 8;
    *(int4*)(As + m * LDS_STRIDE + k8) = ra;
    const int k = t >> 3, n8 = (t & 7) * 8;
    union { int4 v; _Float16 h[8]; } u; u.v = rb;
#pragma unroll
    for (int q = 0; q < 8; ++q) Bs[(n8 + q) * LDS_STRIDE + k] = u.h[q];
  };

  load_global(0);
  for (int kk = 0; kk < K; kk += 32) {
    store_lds();
    __syncthreads();
    if (kk + 32 < K) load_global(kk + 32);

    const v16h bf = load_frag(Bs, tn * 16);
    const v16h a0 = load_frag(As, tmA * 16);
    const v16h a1 = load_frag(As, (tmA + 2) * 16);
    acc0 = __builtin_amdgcn_wmma_f32_16x16x32_f16(false, a0, false, bf,
                                                  (short)0, acc0, false, false);
    acc1 = __builtin_amdgcn_wmma_f32_16x16x32_f16(false, a1, false, bf,
                                                  (short)0, acc1, false, false);
    __syncthreads();
  }

  const int nn = n0 + tn * 16 + (lane & 15);
  const int hh = (lane >> 4) * 8;
  float sc = 1.0f;
  if (OUT_F32 && scale) sc = scale[g];
#pragma unroll
  for (int which = 0; which < 2; ++which) {
    const v8f acc = which ? acc1 : acc0;
    const int mbase = m0 + (which ? (tmA + 2) : tmA) * 16 + hh;
#pragma unroll
    for (int r = 0; r < 8; ++r) {
      const size_t idx = (size_t)(mbase + r) * ldd + nn;
      if (OUT_F32) {
        ((float*)Dv)[(size_t)g * sD + idx] = acc[r] * sc;
      } else {
        float v = alpha * acc[r];
        if (Cadd) v += beta * (float)Cadd[idx];
        ((_Float16*)Dv)[(size_t)g * sD + idx] = (_Float16)v;
      }
    }
  }
}

// ---- row-center (mean over fan_in=512) and convert to f16; 1 wave32 per row
__global__ __launch_bounds__(256) void center_to_f16(
    const float* __restrict__ W, _Float16* __restrict__ Zc) {
  const int row = blockIdx.x * 8 + (threadIdx.x >> 5);
  const int lane = threadIdx.x & 31;
  union { float4 f4[4]; float f[16]; } in;
  const float4* rp = (const float4*)(W + (size_t)row * 512 + lane * 16);
#pragma unroll
  for (int j = 0; j < 4; ++j) in.f4[j] = rp[j];
  float s = 0.0f;
#pragma unroll
  for (int j = 0; j < 16; ++j) s += in.f[j];
#pragma unroll
  for (int off = 16; off >= 1; off >>= 1) s += __shfl_xor(s, off, 32);
  const float mean = s * (1.0f / 512.0f);
  union { int4 q[2]; _Float16 h[16]; } o;
#pragma unroll
  for (int j = 0; j < 16; ++j) o.h[j] = (_Float16)(in.f[j] - mean);
  int4* op = (int4*)(Zc + (size_t)row * 512 + lane * 16);
  op[0] = o.q[0];
  op[1] = o.q[1];
}

// ---- S^2 partial reduction (deterministic) + S -> f16 convert
__global__ __launch_bounds__(256) void square_convert_reduce(
    const float* __restrict__ S, _Float16* __restrict__ Sh,
    float* __restrict__ partials) {
  __shared__ float red[8];
  const size_t i = (size_t)blockIdx.x * 256 + threadIdx.x;
  const float s = S[i];
  Sh[i] = (_Float16)s;
  float v = s * s;
#pragma unroll
  for (int off = 16; off >= 1; off >>= 1) v += __shfl_xor(v, off, 32);
  if ((threadIdx.x & 31) == 0) red[threadIdx.x >> 5] = v;
  __syncthreads();
  if (threadIdx.x == 0) {
    float x = 0.0f;
#pragma unroll
    for (int j = 0; j < 8; ++j) x += red[j];
    partials[blockIdx.x] = x;
  }
}

// ---- per-group final reduce: scale[g] = (sum S^2)^(-1/4) = 1/sqrt(||S||_F)
__global__ __launch_bounds__(256) void finalize_scales(
    const float* __restrict__ partials, float* __restrict__ scales) {
  __shared__ float red[256];
  const int g = blockIdx.x;
  float v = 0.0f;
  for (int j = threadIdx.x; j < 1024; j += 256) v += partials[g * 1024 + j];
  red[threadIdx.x] = v;
  __syncthreads();
  for (int s = 128; s > 0; s >>= 1) {
    if (threadIdx.x < s) red[threadIdx.x] += red[threadIdx.x + s];
    __syncthreads();
  }
  if (threadIdx.x == 0) scales[g] = powf(red[0], -0.25f);
}

// ---- B0 = I (f16), batched over groups
__global__ __launch_bounds__(256) void init_identity_f16(_Float16* __restrict__ B) {
  const size_t i = (size_t)blockIdx.x * 256 + threadIdx.x;
  const int rem = (int)(i & (512 * 512 - 1));
  const int r = rem >> 9, c = rem & 511;
  B[i] = (r == c) ? (_Float16)1.0f : (_Float16)0.0f;
}

extern "C" void kernel_launch(void* const* d_in, const int* in_sizes, int n_in,
                              void* d_out, int out_size, void* d_ws, size_t ws_size,
                              hipStream_t stream) {
  (void)in_sizes; (void)n_in; (void)out_size; (void)ws_size;
  const float* W = (const float*)d_in[0];
  float* out = (float*)d_out;

  const size_t ZC = (size_t)65536 * 512;
  const size_t SE = (size_t)4 * 512 * 512;

  char* p = (char*)d_ws;
  auto take = [&](size_t bytes) -> char* {
    char* q = p;
    p += (bytes + 255) & ~(size_t)255;
    return q;
  };
  _Float16* Zc  = (_Float16*)take(ZC * 2);   // natural centered f16
  _Float16* ZcB = (_Float16*)take(ZC * 2);   // fragment-packed (B-form)
  float*    Sf  = (float*)take(SE * 4);
  _Float16* Sh  = (_Float16*)take(SE * 2);
  _Float16* Bb0 = (_Float16*)take(SE * 2);
  _Float16* Bb1 = (_Float16*)take(SE * 2);
  _Float16* P1  = (_Float16*)take(SE * 2);
  _Float16* P2  = (_Float16*)take(SE * 2);
  float* partials = (float*)take(4096 * 4);
  float* scales   = (float*)take(4 * 4);

  const long long sZ = 16384LL * 512;
  const long long sS = 512LL * 512;

  // 1) center + f16 ; 2) pack to fragment order
  center_to_f16<<<8192, 256, 0, stream>>>(W, Zc);
  pack_zc_bform<<<8192, 256, 0, stream>>>(Zc, ZcB);

  // 3) S = Zc^T Zc from packed operands (M=N=512, K=16384)
  syrk_packed<<<dim3(4, 4, 4), 256, 0, stream>>>(ZcB, Sf);

  // 4) ||S||_F^2 partials + S->f16 ; scales[g] = (sum)^(-1/4)
  square_convert_reduce<<<4096, 256, 0, stream>>>(Sf, Sh, partials);
  finalize_scales<<<4, 256, 0, stream>>>(partials, scales);

  // 5) B = I; 5 Newton-Schulz iterations: B <- 1.5 B - 0.5 (B*B*B)*S
  init_identity_f16<<<4096, 256, 0, stream>>>(Bb0);
  _Float16 *Bcur = Bb0, *Bnext = Bb1;
  for (int it = 0; it < 5; ++it) {
    gemm_wmma<false><<<dim3(8, 8, 4), 256, 0, stream>>>(
        Bcur, Bcur, nullptr, P1, 512, sS, sS, 0, sS, 512, 512, 512, 1.0f, 0.0f,
        nullptr);
    gemm_wmma<false><<<dim3(8, 8, 4), 256, 0, stream>>>(
        P1, Bcur, nullptr, P2, 512, sS, sS, 0, sS, 512, 512, 512, 1.0f, 0.0f,
        nullptr);
    gemm_wmma<false><<<dim3(8, 8, 4), 256, 0, stream>>>(
        P2, Sh, Bcur, Bnext, 512, sS, sS, sS, sS, 512, 512, 512, -0.5f, 1.5f,
        nullptr);
    _Float16* tmp = Bcur; Bcur = Bnext; Bnext = tmp;
  }

  // 6) W = (Zc @ B) * scale[g]  (M=16384/group, N=512, K=512), f32 out
  gemm_wmma<true><<<dim3(8, 256, 4), 256, 0, stream>>>(
      Zc, Bcur, nullptr, out, 512, sZ, sS, 0, sZ, 512, 512, 512, 1.0f, 0.0f,
      scales);
}
